// SSNet_50663434223895
// MI455X (gfx1250) — compile-verified, matrix operand order checked by reading, not run on hardware
//
#include <hip/hip_runtime.h>
#include <hip/hip_bf16.h>

typedef __attribute__((ext_vector_type(16))) _Float16 v16h;
typedef __attribute__((ext_vector_type(8)))  _Float16 v8h;
typedef __attribute__((ext_vector_type(8)))  float    v8f;

// Problem constants (match reference)
constexpr int kB   = 16384;
constexpr int kT   = 32;
constexpr int kIN  = 64;
constexpr int kH   = 512;
constexpr int kG   = 2048;   // 4*H
constexpr int kOUT = 19;

// Tiling
constexpr int kMB  = 64;        // batch rows per block (4 WMMA M-tiles)
constexpr int kMT  = 4;         // M-tiles per wave
constexpr int kTH  = 256;       // 8 waves
constexpr int kNT  = kG / 16;   // 128 N-tiles of 16 gate columns

// LDS padded row strides (in halves) for bank-conflict-free ds_load_b128
constexpr int kHP  = 520;       // 512 + 8  -> 1040 B rows, rotates 4 banks/row
constexpr int kINP = 72;        // 64 + 8   -> 144 B rows

__device__ __forceinline__ float sigf(float x)  { return 1.0f / (1.0f + __expf(-x)); }
__device__ __forceinline__ float tanhn(float x) { return 2.0f / (1.0f + __expf(-2.0f * x)) - 1.0f; }

__device__ __forceinline__ v16h wmma_a_from_lds(const _Float16* row, int koff, int lhi) {
    // 16-bit A 16x32 layout: lanes 0-15 hold K halves 0..7,16..23, lanes 16-31 hold 8..15,24..31
    const _Float16* p0 = row + koff + lhi * 8;
    v8h lo = *(const v8h*)(p0);
    v8h hi = *(const v8h*)(p0 + 16);
    return __builtin_shufflevector(lo, hi, 0,1,2,3,4,5,6,7,8,9,10,11,12,13,14,15);
}

// ---------------------------------------------------------------------------
// Pre-pass: pack a (krows x 2048) fp32 weight matrix into f16 WMMA B-fragments.
// Fragment (kt, nt): 32 lanes x 16 halves, per-lane contiguous (32B).
// ---------------------------------------------------------------------------
__global__ void pack_frags(const float* __restrict__ src, _Float16* __restrict__ dst, int total) {
    int e = blockIdx.x * blockDim.x + threadIdx.x;
    if (e >= total) return;
    int s    = e & 15;
    int l    = (e >> 4) & 31;
    int f    = e >> 9;
    int kt   = f >> 7;      // / 128
    int nt   = f & 127;
    int v    = s >> 1;
    int hh   = s & 1;
    int lhi  = l >> 4;
    int lmod = l & 15;
    int kl   = (v < 4) ? (lhi * 8 + v * 2 + hh) : (16 + lhi * 8 + (v - 4) * 2 + hh);
    int k    = kt * 32 + kl;
    int n    = nt * 16 + lmod;
    dst[e] = (_Float16)src[(size_t)k * kG + n];
}

// ---------------------------------------------------------------------------
// Fused 2-layer LSTM: each block owns kMB batch rows for all T steps.
// h kept in LDS (f16, double-buffered); c kept in global f32 (L2-resident),
// each (m,n) cell touched by exactly one lane. Accumulators start as inline-0
// C operands; bias / flag rank-1 terms are folded into the gate epilogue.
// ---------------------------------------------------------------------------
__global__ __launch_bounds__(kTH)
void lstm_fused(const float* __restrict__ xin,
                const _Float16* __restrict__ wx0f, const _Float16* __restrict__ wh0f,
                const _Float16* __restrict__ wx1f, const _Float16* __restrict__ wh1f,
                const float* __restrict__ b0, const float* __restrict__ b1,
                const float* __restrict__ w1row0,
                const float* __restrict__ wlin, const float* __restrict__ blin,
                float* __restrict__ c0g, float* __restrict__ c1g,
                float* __restrict__ out)
{
    extern __shared__ char smem[];
    _Float16* h0buf = (_Float16*)smem;                    // 2 * kMB*kHP halves
    _Float16* h1buf = h0buf + 2 * kMB * kHP;              // 2 * kMB*kHP halves
    _Float16* xs    = h1buf + 2 * kMB * kHP;              // kMB * kINP halves
    float*    flg   = (float*)(xs + kMB * kINP);          // kMB floats

    const int tid   = threadIdx.x;
    const int lane  = tid & 31;
    const int wv    = tid >> 5;
    const int lmod  = lane & 15;
    const int lhi   = lane >> 4;
    const int brow  = blockIdx.x * kMB;

    // zero-init h state (c handled by t==0 special case)
    for (int e = tid; e < kMB * kHP; e += kTH) {
        h0buf[e] = (_Float16)0.f; h0buf[kMB * kHP + e] = (_Float16)0.f;
        h1buf[e] = (_Float16)0.f; h1buf[kMB * kHP + e] = (_Float16)0.f;
    }
    __syncthreads();

    int cur = 0, nxt = 1;

    for (int t = 0; t < kT; ++t) {
        // ---- stage x_t tile to LDS (f16) + flag ----
        for (int e = tid; e < kMB * kIN; e += kTH) {
            int m = e >> 6;       // / kIN
            int i = e & 63;
            float v = xin[(size_t)(brow + m) * (kT * kIN) + (size_t)t * kIN + i];
            xs[m * kINP + i] = (_Float16)v;
            if (i == 0) flg[m] = v;
        }
        __syncthreads();

        const _Float16* h0c = h0buf + cur * kMB * kHP;
        _Float16*       h0n = h0buf + nxt * kMB * kHP;
        const _Float16* h1c = h1buf + cur * kMB * kHP;
        _Float16*       h1n = h1buf + nxt * kMB * kHP;

        // ================= Layer 0 =================
        for (int hti = 0; hti < 4; ++hti) {
            const int ht = wv * 4 + hti;         // hidden-unit tile 0..31
            const int nh = ht * 16 + lmod;       // gate column within one gate block
            v8f acc[kMT][4];
            const v8f zero{};
            #pragma unroll
            for (int gt = 0; gt < 4; ++gt)
                #pragma unroll
                for (int mt = 0; mt < kMT; ++mt) acc[mt][gt] = zero;

            // x @ Wx0 (K = 64 -> 2 k-tiles); first group consumes inline-0 C
            #pragma unroll
            for (int kt = 0; kt < 2; ++kt) {
                v16h a[kMT];
                #pragma unroll
                for (int mt = 0; mt < kMT; ++mt)
                    a[mt] = wmma_a_from_lds(xs + (mt * 16 + lmod) * kINP, kt * 32, lhi);
                #pragma unroll
                for (int gt = 0; gt < 4; ++gt) {
                    const _Float16* bp = wx0f + ((size_t)(kt * kNT + gt * 32 + ht)) * 512 + lane * 16;
                    v16h bb = *(const v16h*)bp;
                    #pragma unroll
                    for (int mt = 0; mt < kMT; ++mt)
                        acc[mt][gt] = __builtin_amdgcn_wmma_f32_16x16x32_f16(false, a[mt], false, bb, (short)0, acc[mt][gt], false, false);
                }
            }
            // h0 @ Wh0 (K = 512 -> 16 k-tiles)
            for (int kt = 0; kt < 16; ++kt) {
                v16h a[kMT];
                #pragma unroll
                for (int mt = 0; mt < kMT; ++mt)
                    a[mt] = wmma_a_from_lds(h0c + (mt * 16 + lmod) * kHP, kt * 32, lhi);
                #pragma unroll
                for (int gt = 0; gt < 4; ++gt) {
                    const _Float16* bp = wh0f + ((size_t)(kt * kNT + gt * 32 + ht)) * 512 + lane * 16;
                    v16h bb = *(const v16h*)bp;
                    #pragma unroll
                    for (int mt = 0; mt < kMT; ++mt)
                        acc[mt][gt] = __builtin_amdgcn_wmma_f32_16x16x32_f16(false, a[mt], false, bb, (short)0, acc[mt][gt], false, false);
                }
            }
            // gate epilogue: bias + nonlinearity + flag blend
            const float bi = b0[nh],          bf = b0[kH + nh];
            const float bg = b0[2 * kH + nh], bo = b0[3 * kH + nh];
            #pragma unroll
            for (int mt = 0; mt < kMT; ++mt) {
                #pragma unroll
                for (int v = 0; v < 8; ++v) {
                    int m = mt * 16 + lhi * 8 + v;
                    size_t cidx = (size_t)(brow + m) * kH + nh;
                    float gi = acc[mt][0][v] + bi, gf = acc[mt][1][v] + bf;
                    float gg = acc[mt][2][v] + bg, go = acc[mt][3][v] + bo;
                    float cold = (t == 0) ? 0.f : c0g[cidx];
                    float cn = sigf(gf) * cold + sigf(gi) * tanhn(gg);
                    float hn = sigf(go) * tanhn(cn);
                    float fl = flg[m];
                    float cb = cold + fl * (cn - cold);
                    float hold = (float)h0c[m * kHP + nh];
                    float hb = hold + fl * (hn - hold);
                    c0g[cidx] = cb;
                    h0n[m * kHP + nh] = (_Float16)hb;
                }
            }
        }
        __syncthreads();   // h0n fully written before layer 1 reads it

        // ================= Layer 1 =================
        for (int hti = 0; hti < 4; ++hti) {
            const int ht = wv * 4 + hti;
            const int nh = ht * 16 + lmod;
            v8f acc[kMT][4];
            const v8f zero{};

            // h0_new @ Wx1[1:,:]  (K = 512); kt = 0 peeled for inline-0 C
            {
                v16h a[kMT];
                #pragma unroll
                for (int mt = 0; mt < kMT; ++mt)
                    a[mt] = wmma_a_from_lds(h0n + (mt * 16 + lmod) * kHP, 0, lhi);
                #pragma unroll
                for (int gt = 0; gt < 4; ++gt) {
                    const _Float16* bp = wx1f + ((size_t)(gt * 32 + ht)) * 512 + lane * 16;
                    v16h bb = *(const v16h*)bp;
                    #pragma unroll
                    for (int mt = 0; mt < kMT; ++mt)
                        acc[mt][gt] = __builtin_amdgcn_wmma_f32_16x16x32_f16(false, a[mt], false, bb, (short)0, zero, false, false);
                }
            }
            for (int kt = 1; kt < 16; ++kt) {
                v16h a[kMT];
                #pragma unroll
                for (int mt = 0; mt < kMT; ++mt)
                    a[mt] = wmma_a_from_lds(h0n + (mt * 16 + lmod) * kHP, kt * 32, lhi);
                #pragma unroll
                for (int gt = 0; gt < 4; ++gt) {
                    const _Float16* bp = wx1f + ((size_t)(kt * kNT + gt * 32 + ht)) * 512 + lane * 16;
                    v16h bb = *(const v16h*)bp;
                    #pragma unroll
                    for (int mt = 0; mt < kMT; ++mt)
                        acc[mt][gt] = __builtin_amdgcn_wmma_f32_16x16x32_f16(false, a[mt], false, bb, (short)0, acc[mt][gt], false, false);
                }
            }
            // h1 @ Wh1 (K = 512)
            for (int kt = 0; kt < 16; ++kt) {
                v16h a[kMT];
                #pragma unroll
                for (int mt = 0; mt < kMT; ++mt)
                    a[mt] = wmma_a_from_lds(h1c + (mt * 16 + lmod) * kHP, kt * 32, lhi);
                #pragma unroll
                for (int gt = 0; gt < 4; ++gt) {
                    const _Float16* bp = wh1f + ((size_t)(kt * kNT + gt * 32 + ht)) * 512 + lane * 16;
                    v16h bb = *(const v16h*)bp;
                    #pragma unroll
                    for (int mt = 0; mt < kMT; ++mt)
                        acc[mt][gt] = __builtin_amdgcn_wmma_f32_16x16x32_f16(false, a[mt], false, bb, (short)0, acc[mt][gt], false, false);
                }
            }
            // gate epilogue: bias + flag rank-1 term + nonlinearity + blend
            const float bi = b1[nh],          bf = b1[kH + nh];
            const float bg = b1[2 * kH + nh], bo = b1[3 * kH + nh];
            const float wi = w1row0[nh],          wf = w1row0[kH + nh];
            const float wg = w1row0[2 * kH + nh], wo = w1row0[3 * kH + nh];
            #pragma unroll
            for (int mt = 0; mt < kMT; ++mt) {
                #pragma unroll
                for (int v = 0; v < 8; ++v) {
                    int m = mt * 16 + lhi * 8 + v;
                    size_t cidx = (size_t)(brow + m) * kH + nh;
                    float fl = flg[m];
                    float gi = acc[mt][0][v] + bi + fl * wi;
                    float gf = acc[mt][1][v] + bf + fl * wf;
                    float gg = acc[mt][2][v] + bg + fl * wg;
                    float go = acc[mt][3][v] + bo + fl * wo;
                    float cold = (t == 0) ? 0.f : c1g[cidx];
                    float cn = sigf(gf) * cold + sigf(gi) * tanhn(gg);
                    float hn = sigf(go) * tanhn(cn);
                    float cb = cold + fl * (cn - cold);
                    float hold = (float)h1c[m * kHP + nh];
                    float hb = hold + fl * (hn - hold);
                    c1g[cidx] = cb;
                    h1n[m * kHP + nh] = (_Float16)hb;
                }
            }
        }
        __syncthreads();   // h1n complete; also protects xs/h0 for next step

        int tmp = cur; cur = nxt; nxt = tmp;
    }

    // ---- final projection: out = h1 @ Wlin + blin  (512x19, negligible) ----
    const _Float16* hfin = h1buf + cur * kMB * kHP;
    for (int e = tid; e < kMB * kOUT; e += kTH) {
        int m = e / kOUT;
        int j = e % kOUT;
        const _Float16* hp = hfin + m * kHP;
        float s = blin[j];
        for (int k = 0; k < kH; ++k) s += (float)hp[k] * wlin[k * kOUT + j];
        out[(size_t)(brow + m) * kOUT + j] = s;
    }
}

extern "C" void kernel_launch(void* const* d_in, const int* in_sizes, int n_in,
                              void* d_out, int out_size, void* d_ws, size_t ws_size,
                              hipStream_t stream) {
    const float* xin  = (const float*)d_in[0];
    const float* Wx0  = (const float*)d_in[1];
    const float* Wh0  = (const float*)d_in[2];
    const float* b0   = (const float*)d_in[3];
    const float* Wx1  = (const float*)d_in[4];   // (513, 2048): row 0 = flag weights
    const float* Wh1  = (const float*)d_in[5];
    const float* b1   = (const float*)d_in[6];
    const float* Wlin = (const float*)d_in[7];
    const float* blin = (const float*)d_in[8];
    float* out        = (float*)d_out;

    // workspace layout: packed f16 B-fragments, then f32 cell states (L2-resident)
    _Float16* ws   = (_Float16*)d_ws;
    _Float16* wx0f = ws;                         //  64*2048 halves
    _Float16* wh0f = wx0f + (size_t)kIN * kG;    // 512*2048 halves
    _Float16* wx1f = wh0f + (size_t)kH * kG;     // 512*2048 halves (rows 1..512 of Wx1)
    _Float16* wh1f = wx1f + (size_t)kH * kG;     // 512*2048 halves
    float*    c0g  = (float*)(wh1f + (size_t)kH * kG);   // kB*kH f32
    float*    c1g  = c0g + (size_t)kB * kH;              // kB*kH f32

    {
        int n0 = kIN * kG;
        pack_frags<<<(n0 + 255) / 256, 256, 0, stream>>>(Wx0, wx0f, n0);
        int n1 = kH * kG;
        pack_frags<<<(n1 + 255) / 256, 256, 0, stream>>>(Wh0, wh0f, n1);
        pack_frags<<<(n1 + 255) / 256, 256, 0, stream>>>(Wx1 + kG, wx1f, n1);
        pack_frags<<<(n1 + 255) / 256, 256, 0, stream>>>(Wh1, wh1f, n1);
    }

    // dynamic LDS: 2x h0(f16) + 2x h1(f16) + x(f16) + flag(f32), padded rows
    size_t smem = (size_t)(2 * kMB * kHP * 2) * 2   // h0buf + h1buf (double buffered)
                + (size_t)(kMB * kINP * 2)          // xs
                + (size_t)(kMB * 4);                // flg
    hipFuncSetAttribute((const void*)lstm_fused,
                        hipFuncAttributeMaxDynamicSharedMemorySize, (int)smem);

    lstm_fused<<<kB / kMB, kTH, smem, stream>>>(xin, wx0f, wh0f, wx1f, wh1f,
                                                b0, b1, Wx1 /* row 0 */, Wlin, blin,
                                                c0g, c1g, out);
}
